// GRU4RecUserModule_82703890252105
// MI455X (gfx1250) — matrix operand order repaired and checked by reading, not compiled, not run on hardware
//
#include <hip/hip_runtime.h>
#include <hip/hip_bf16.h>

typedef __attribute__((ext_vector_type(16))) __bf16 v16bf;
typedef __attribute__((ext_vector_type(8)))  __bf16 bf16x8;
typedef __attribute__((ext_vector_type(8)))  float  v8f;
typedef __attribute__((ext_vector_type(4)))  float  f32x4;

#define HDIM 128
#define NORM_EPS 1e-12f

__device__ __forceinline__ __bf16 f2bf(float f) { return (__bf16)f; }

__device__ __forceinline__ float fsig(float x) {
  return 1.0f / (1.0f + __expf(-x));
}
__device__ __forceinline__ float ftanh(float x) {
  float e = __expf(-2.0f * x);
  return (1.0f - e) / (1.0f + e);
}

// ---- CDNA5 async global->LDS staging (ISA 10.x / CDNA5 ISA) ----
// Copies 32 bytes per lane: LDS[lds_off .. +31] = MEM[gptr .. +31], tracked
// by ASYNCcnt (2 instructions -> +2 on this wave's counter).
__device__ __forceinline__ void async_load_32B(unsigned lds_off, const float* gptr) {
  unsigned long long ga = (unsigned long long)(uintptr_t)gptr;
  asm volatile(
      "global_load_async_to_lds_b128 %0, %1, off\n\t"
      "global_load_async_to_lds_b128 %0, %1, off offset:16"
      :: "v"(lds_off), "v"(ga) : "memory");
}
__device__ __forceinline__ void wait_async_le2() {
  asm volatile("s_wait_asynccnt 0x2" ::: "memory");
}
__device__ __forceinline__ void wait_async_0() {
  asm volatile("s_wait_asynccnt 0x0" ::: "memory");
}

// 16x32 bf16 A-fragment (M rows, K cols) from row-major [16][128] LDS tile.
// CDNA5 ISA 7.12.2 (16-bit A 16x32): m = lane&15, hi = lane>>4;
// elems 0..7 -> K = k0+8*hi+0..7 ; elems 8..15 -> K = k0+16+8*hi+0..7.
__device__ __forceinline__ v16bf load_a_frag(const __bf16* lds, int k0) {
  const int lane = threadIdx.x & 31;
  const int m = lane & 15, hi = lane >> 4;
  const __bf16* p = lds + m * HDIM + k0 + hi * 8;
  bf16x8 lo  = *(const bf16x8*)p;
  bf16x8 hi8 = *(const bf16x8*)(p + 16);
  v16bf out;
#pragma unroll
  for (int i = 0; i < 8; ++i) { out[i] = lo[i]; out[8 + i] = hi8[i]; }
  return out;
}

// 32x16 bf16 B-fragment for B = W^T (B[k][n] = W[n][k], W row-major ld=ldw).
// CDNA5 16-bit B: n = lane&15, hi = lane>>4; elems 0..7 -> K = k0+8*hi+0..7;
// elems 8..15 -> K = k0+16+8*hi+0..7, packed 2 per VGPR.
__device__ __forceinline__ v16bf load_bT_frag(const float* W, int ldw, int nbase, int k0) {
  const int lane = threadIdx.x & 31;
  const int n = lane & 15, hi = lane >> 4;
  const float* p = W + (size_t)(nbase + n) * ldw + k0 + hi * 8;
  v16bf out;
#pragma unroll
  for (int i = 0; i < 16; ++i) out[i] = f2bf(p[i < 8 ? i : i + 8]);
  return out;
}

// ---------------------------------------------------------------------------
// Prep: deterministic counting sort of sequences by length (descending).
// Tiling similar-length sequences together cuts padded recurrence work ~1.9x.
// Rank within equal lengths = original index order -> fully deterministic.
// ---------------------------------------------------------------------------
__global__ __launch_bounds__(256)
void prep_perm_kernel(const int* __restrict__ offset, int* __restrict__ perm,
                      int B, int T) {
  __shared__ int s_len[4096];
  __shared__ int s_cnt[256];
  __shared__ int s_start[256];
  const int tid = threadIdx.x;

  if (B > 4096) {  // fallback: identity
    for (int i = tid; i < B; i += 256) perm[i] = i;
    return;
  }
  for (int i = tid; i < B; i += 256) {
    int o = offset[i];
    int e = (i + 1 < B) ? offset[i + 1] : T;
    s_len[i] = e - o;
  }
  s_cnt[tid] = 0;
  __syncthreads();
  for (int i = tid; i < B; i += 256) {
    int bin = 255 - min(s_len[i], 255);   // descending by length
    atomicAdd(&s_cnt[bin], 1);
  }
  __syncthreads();
  if (tid == 0) {
    int acc = 0;
    for (int b = 0; b < 256; ++b) { s_start[b] = acc; acc += s_cnt[b]; }
  }
  __syncthreads();
  for (int i = tid; i < B; i += 256) {
    int L = min(s_len[i], 255);
    int r = 0;
    for (int j = 0; j < i; ++j) r += (min(s_len[j], 255) == L) ? 1 : 0;
    perm[s_start[255 - L] + r] = i;
  }
}

// ---------------------------------------------------------------------------
// Fused GRU + dense + L2-normalize. One workgroup = 16 (length-sorted)
// sequences; 8 waves each own a 16-column slice of H across all 3 gates.
// ---------------------------------------------------------------------------
__global__ __launch_bounds__(256)
void gru4rec_fused_kernel(const float* __restrict__ x,
                          const int*   __restrict__ offset,
                          const float* __restrict__ W_ih,
                          const float* __restrict__ W_hh,
                          const float* __restrict__ W_dense,
                          const float* __restrict__ b_dense,
                          const int*   __restrict__ perm,
                          float* __restrict__ out,
                          int B, int T)
{
  __shared__ __align__(16) float  s_xf[2][16 * HDIM]; // async-staged x_t (f32, dbl buf)
  __shared__ __align__(16) __bf16 s_x16[16 * HDIM];   // x_t tile, bf16 (WMMA A)
  __shared__ __align__(16) __bf16 s_h16[16 * HDIM];   // h state, bf16 (WMMA A)
  __shared__ __align__(16) float  s_hf [16 * HDIM];   // h state, f32 (z*h blend)
  __shared__ __align__(16) float  s_fin[16 * HDIM];   // captured h at len-1
  __shared__ int   s_off[16];
  __shared__ int   s_len[16];
  __shared__ int   s_seq[16];
  __shared__ float s_nrm[16];

  const int tid  = threadIdx.x;
  const int seq0 = blockIdx.x * 16;

  if (tid < 16) {
    int slot = seq0 + tid;
    int s = (slot < B) ? perm[slot] : -1;
    int o = (s >= 0) ? offset[s] : 0;
    int e = (s >= 0) ? ((s + 1 < B) ? offset[s + 1] : T) : 1;
    s_seq[tid] = s;
    s_off[tid] = o;
    s_len[tid] = (s >= 0) ? (e - o) : 1;
    s_nrm[tid] = 0.0f;
  }
  for (int i = tid; i < 16 * HDIM; i += 256) {
    s_h16[i] = f2bf(0.0f);
    s_hf[i]  = 0.0f;
    s_fin[i] = 0.0f;
  }
  __syncthreads();

  int maxlen = 0;
#pragma unroll
  for (int i = 0; i < 16; ++i) maxlen = max(maxlen, s_len[i]);

  // Per-wave geometry: wave w owns hidden cols [16w, 16w+16).
  const int wave = tid >> 5;
  const int lane = tid & 31;
  const int col  = wave * 16 + (lane & 15); // N in C/D layout
  const int mhi  = (lane >> 4) * 8;         // M base in C/D layout

  int lenM[8];
#pragma unroll
  for (int v = 0; v < 8; ++v) lenM[v] = s_len[mhi + v];

  // Preload all B-fragments (gates r/z/n of W_ih and W_hh) into registers.
  v16bf Bih[3][4], Bhh[3][4];
#pragma unroll
  for (int g = 0; g < 3; ++g)
#pragma unroll
    for (int c = 0; c < 4; ++c) {
      Bih[g][c] = load_bT_frag(W_ih, HDIM, g * HDIM + wave * 16, c * 32);
      Bhh[g][c] = load_bT_frag(W_hh, HDIM, g * HDIM + wave * 16, c * 32);
    }

  // x staging geometry: thread -> (row, 8-float segment); each thread
  // async-copies and later converts exactly its own 32 bytes.
  const int xrow = tid >> 4;
  const int xcg  = (tid & 15) * 8;
  const int rlen = s_len[xrow];
  const int roff = s_off[xrow];

  // Prologue: stage x_0 into buffer 0.
  {
    int tc = min(0, rlen - 1);
    const float* px = x + (size_t)(roff + tc) * HDIM + xcg;
    unsigned lds = (unsigned)(uintptr_t)&s_xf[0][xrow * HDIM + xcg];
    async_load_32B(lds, px);
  }

  for (int t = 0; t < maxlen; ++t) {
    // ---- issue async stage of x_{t+1}, then wait for x_t ----
    if (t + 1 < maxlen) {
      int tc = min(t + 1, rlen - 1);  // clamp: stay inside this sequence
      const float* px = x + (size_t)(roff + tc) * HDIM + xcg;
      unsigned lds = (unsigned)(uintptr_t)&s_xf[(t + 1) & 1][xrow * HDIM + xcg];
      async_load_32B(lds, px);
      wait_async_le2();               // only x_t's pair must have landed
    } else {
      wait_async_0();
    }

    // ---- convert own 32B of x_t to bf16 (zero past end of sequence) ----
    {
      union { __bf16 e[8]; bf16x8 v; } u;
      if (t < rlen) {
        const float* pf = &s_xf[t & 1][xrow * HDIM + xcg];
        f32x4 a = *(const f32x4*)pf;
        f32x4 b = *(const f32x4*)(pf + 4);
        u.e[0]=f2bf(a.x); u.e[1]=f2bf(a.y); u.e[2]=f2bf(a.z); u.e[3]=f2bf(a.w);
        u.e[4]=f2bf(b.x); u.e[5]=f2bf(b.y); u.e[6]=f2bf(b.z); u.e[7]=f2bf(b.w);
      } else {
#pragma unroll
        for (int i = 0; i < 8; ++i) u.e[i] = f2bf(0.0f);
      }
      *(bf16x8*)&s_x16[xrow * HDIM + xcg] = u.v;
    }
    __syncthreads();   // B1: x_t (bf16) + h stores from prev iter visible

    // ---- WMMA: split x-part / h-part accumulators (chain depth 4) ----
    v8f aRX = {0,0,0,0,0,0,0,0}, aRH = {0,0,0,0,0,0,0,0};
    v8f aZX = {0,0,0,0,0,0,0,0}, aZH = {0,0,0,0,0,0,0,0};
    v8f aNX = {0,0,0,0,0,0,0,0}, aNH = {0,0,0,0,0,0,0,0};
#pragma unroll
    for (int c = 0; c < 4; ++c) {
      v16bf Ax = load_a_frag(s_x16, c * 32);
      v16bf Ah = load_a_frag(s_h16, c * 32);
      aRX = __builtin_amdgcn_wmma_f32_16x16x32_bf16(false, Ax, false, Bih[0][c], (short)0, aRX, false, false);
      aRH = __builtin_amdgcn_wmma_f32_16x16x32_bf16(false, Ah, false, Bhh[0][c], (short)0, aRH, false, false);
      aZX = __builtin_amdgcn_wmma_f32_16x16x32_bf16(false, Ax, false, Bih[1][c], (short)0, aZX, false, false);
      aZH = __builtin_amdgcn_wmma_f32_16x16x32_bf16(false, Ah, false, Bhh[1][c], (short)0, aZH, false, false);
      aNX = __builtin_amdgcn_wmma_f32_16x16x32_bf16(false, Ax, false, Bih[2][c], (short)0, aNX, false, false);
      aNH = __builtin_amdgcn_wmma_f32_16x16x32_bf16(false, Ah, false, Bhh[2][c], (short)0, aNH, false, false);
    }

    // ---- pointwise GRU update in C/D layout (lane owns col, rows mhi..+7)
    float hnew[8];
#pragma unroll
    for (int v = 0; v < 8; ++v) {
      float r  = fsig(aRX[v] + aRH[v]);
      float z  = fsig(aZX[v] + aZH[v]);
      float n  = ftanh(aNX[v] + r * aNH[v]);
      float hp = s_hf[(mhi + v) * HDIM + col];
      hnew[v] = (1.0f - z) * n + z * hp;
    }
    __syncthreads();   // B2: all waves' A-fragment reads done before h update
#pragma unroll
    for (int v = 0; v < 8; ++v) {
      int m = mhi + v;
      s_hf [m * HDIM + col] = hnew[v];
      s_h16[m * HDIM + col] = f2bf(hnew[v]);
      if (t == lenM[v] - 1) s_fin[m * HDIM + col] = hnew[v];
    }
  }
  __syncthreads();

  // ---- dense (128x128) + L2 normalize + scatter to original rows ------
  {
    const int row = tid >> 4;
    const int j0  = (tid & 15) * 8;
    float acc[8];
#pragma unroll
    for (int u = 0; u < 8; ++u) acc[u] = b_dense[j0 + u];
#pragma unroll
    for (int u = 0; u < 8; ++u) {
      const float* wrow = W_dense + (size_t)(j0 + u) * HDIM;
      float a = acc[u];
      for (int k = 0; k < HDIM; k += 4) {
        f32x4 w  = *(const f32x4*)(wrow + k);
        f32x4 h4 = *(const f32x4*)&s_fin[row * HDIM + k];
        a += w.x * h4.x + w.y * h4.y + w.z * h4.z + w.w * h4.w;
      }
      acc[u] = a;
    }
    float ss = 0.0f;
#pragma unroll
    for (int u = 0; u < 8; ++u) ss += acc[u] * acc[u];
    atomicAdd(&s_nrm[row], ss);
    __syncthreads();
    float scl = 1.0f / fmaxf(sqrtf(s_nrm[row]), NORM_EPS);
    int sdst = s_seq[row];
    if (sdst >= 0) {
      float* po = out + (size_t)sdst * HDIM + j0;
      f32x4 o0 = { acc[0] * scl, acc[1] * scl, acc[2] * scl, acc[3] * scl };
      f32x4 o1 = { acc[4] * scl, acc[5] * scl, acc[6] * scl, acc[7] * scl };
      *(f32x4*)po       = o0;
      *(f32x4*)(po + 4) = o1;
    }
  }
}

extern "C" void kernel_launch(void* const* d_in, const int* in_sizes, int n_in,
                              void* d_out, int out_size, void* d_ws, size_t ws_size,
                              hipStream_t stream) {
  const float* x       = (const float*)d_in[0];
  const int*   offset  = (const int*)  d_in[1];
  const float* W_ih    = (const float*)d_in[2];
  const float* W_hh    = (const float*)d_in[3];
  const float* W_dense = (const float*)d_in[4];
  const float* b_dense = (const float*)d_in[5];
  float*       out     = (float*)d_out;
  int*         perm    = (int*)d_ws;

  const int B = in_sizes[1];
  const int T = in_sizes[0] / HDIM;
  const int tiles = (B + 15) / 16;

  prep_perm_kernel<<<1, 256, 0, stream>>>(offset, perm, B, T);
  gru4rec_fused_kernel<<<tiles, 256, 0, stream>>>(
      x, offset, W_ih, W_hh, W_dense, b_dense, perm, out, B, T);
}
